// GRUForecaster_62955630625431
// MI455X (gfx1250) — matrix operand order repaired and checked by reading, not compiled, not run on hardware
//
#include <hip/hip_runtime.h>
#include <hip/hip_bf16.h>

// ---------------------------------------------------------------------------
// GRU forecaster on MI455X (gfx1250), wave32 + v_wmma_f32_16x16x32_bf16.
// B=32, T=2048, I=256, H=512, O=64. Persistent grid (32 WGs x 192 thr),
// global-atomic grid barrier per timestep. Weight B-fragments held in VGPRs
// across the whole recurrence; per-step A operands (h state + x_t slab)
// staged global->LDS with CDNA5 async copies (ASYNCcnt), read as ds_load_b128.
// ---------------------------------------------------------------------------

#define Bc 32
#define Tc 2048
#define Ic 256
#define Hc 512
#define Oc 64
#define NWG 32
#define NTH 192   // 6 waves: (mtile 0/1) x (gate r/z/n)

typedef __attribute__((ext_vector_type(16))) __bf16          v16bf;
typedef __attribute__((ext_vector_type(8)))  float           v8f;
typedef __attribute__((ext_vector_type(8)))  unsigned short  u16x8;
typedef __attribute__((ext_vector_type(16))) unsigned short  u16x16;

__device__ __forceinline__ unsigned short f2bf(float f) {
  union { float f; unsigned u; } v; v.f = f;
  unsigned r = v.u + 0x7FFFu + ((v.u >> 16) & 1u);   // round-to-nearest-even
  return (unsigned short)(r >> 16);
}
__device__ __forceinline__ float bf2f(unsigned short h) {
  union { unsigned u; float f; } v; v.u = ((unsigned)h) << 16; return v.f;
}

// --- CDNA5 async global->LDS copy (16 B per active lane, ASYNCcnt) ---------
__device__ __forceinline__ void async_copy16(unsigned lds_off,
                                             const unsigned short* gptr) {
  asm volatile("global_load_async_to_lds_b128 %0, %1, off"
               :: "v"(lds_off), "v"(gptr) : "memory");
}
__device__ __forceinline__ void wait_async() {
  asm volatile("s_wait_asynccnt 0x0" ::: "memory");
}

// --- A fragment (16x32 bf16) from a row-major LDS slab ---------------------
// ISA 16-bit A layout: lanes 0-15 -> M rows; per lane-half, elems 0..7 hold
// K {hf*8 .. hf*8+7}, elems 8..15 hold K {16+hf*8 ..}. Two 16B ds loads.
// rowp must already point at  slab + m*rstride + hf*8.
__device__ __forceinline__ v16bf load_a_frag(const unsigned short* rowp, int kbase) {
  const u16x8* p = (const u16x8*)(rowp + kbase);
  u16x8 lo = p[0];
  u16x8 hi = p[2];            // +16 elems
  u16x16 u;
#pragma unroll
  for (int e = 0; e < 8; ++e) { u[e] = lo[e]; u[e + 8] = hi[e]; }
  return __builtin_bit_cast(v16bf, u);
}

// --- B fragment (32x16 bf16) from the pre-packed buffer: 32B/lane ----------
__device__ __forceinline__ v16bf load_b_frag(const unsigned short* __restrict__ basep,
                                             int kt) {
  return *(const v16bf*)(basep + (size_t)kt * 512);
}

__device__ __forceinline__ v8f wmma_bf16(v16bf a, v16bf b, v8f c) {
  return __builtin_amdgcn_wmma_f32_16x16x32_bf16(false, a, false, b,
                                                 (short)0, c, false, false);
}

// --- grid-wide barrier (monotonic counter, release/acquire) ----------------
__device__ __forceinline__ void grid_sync(unsigned* bar, unsigned target) {
  __threadfence();
  __syncthreads();
  if (threadIdx.x == 0) {
    __hip_atomic_fetch_add(bar, 1u, __ATOMIC_RELEASE, __HIP_MEMORY_SCOPE_AGENT);
    while (__hip_atomic_load(bar, __ATOMIC_ACQUIRE, __HIP_MEMORY_SCOPE_AGENT) < target) {
      __builtin_amdgcn_s_sleep(1);
    }
  }
  __syncthreads();
}

// ---------------------------------------------------------------------------
// Prep: pack f32 (K x N row-major) weights into bf16 WMMA B-fragments.
// ---------------------------------------------------------------------------
__global__ void pack_b(const float* __restrict__ W, unsigned short* __restrict__ out,
                       int K, int N) {
  const int tid = blockIdx.x * blockDim.x + threadIdx.x;
  const int KT = K >> 5;
  const int NT = N >> 4;
  if (tid >= KT * NT * 32) return;
  const int lane = tid & 31;
  const int f    = tid >> 5;
  const int kt   = f % KT;
  const int nt   = f / KT;
  const int hf   = lane >> 4;
  const int n    = nt * 16 + (lane & 15);
  const float* src = W + (size_t)(kt * 32 + hf * 8) * N + n;
  unsigned short* dst = out + (size_t)tid * 16;
#pragma unroll
  for (int e = 0; e < 16; ++e) dst[e] = f2bf(src[(size_t)(e < 8 ? e : e + 8) * N]);
}

__global__ void f32_to_bf16(const float* __restrict__ in,
                            unsigned short* __restrict__ out, size_t n) {
  size_t i = blockIdx.x * (size_t)blockDim.x + threadIdx.x;
  const size_t stride = (size_t)gridDim.x * blockDim.x;
  for (; i < n; i += stride) out[i] = f2bf(in[i]);
}

// ---------------------------------------------------------------------------
// One GRU layer recurrence (compile-time KTX -> fully unrolled).
// LDS map: [0,32K) hTile (32x512 bf16) ; [32K,64K) xTile (32x(KTX*32) bf16),
// whose first 8 KB is re-used as the sAcc gate-staging area after the GEMM.
// ---------------------------------------------------------------------------
template <int KTX, bool WRITE_OUT>
__device__ __forceinline__ void gru_layer(
    const unsigned short* __restrict__ xbase, size_t xrs,
    const unsigned short* __restrict__ WiP,
    const unsigned short* __restrict__ WhP,
    const float* __restrict__ bi, const float* __restrict__ bhn,
    unsigned short* __restrict__ h0out,
    unsigned short* __restrict__ hbuf0, unsigned short* __restrict__ hbuf1,
    int& pr, unsigned& phase, unsigned* bar,
    unsigned char* smem,
    int tid, int lane, int mtile, int g, int hf, int c0, int wg) {
  const int ntg = g * 32 + wg;              // 16-wide N tile among 3H/16 = 96
  const int m   = mtile * 16 + (lane & 15); // this lane's A row
  const unsigned short* WiB = WiP + ((size_t)(ntg * KTX) * 32 + lane) * 16;
  const unsigned short* WhB = WhP + ((size_t)(ntg * 16)  * 32 + lane) * 16;

  unsigned short* hTile = (unsigned short*)smem;
  unsigned short* xTile = (unsigned short*)(smem + 32768);
  float (*sAcc)[32][16] = (float (*)[32][16])(smem + 32768);  // aliases xTile
  const unsigned hOff = (unsigned)(unsigned long long)hTile;  // LDS byte offset
  const unsigned xOff = (unsigned)(unsigned long long)xTile;

  // ---- hold this wave's weight slice in VGPRs for the whole recurrence ----
  v16bf wiR[KTX], whR[16];
#pragma unroll
  for (int kt = 0; kt < KTX; ++kt) wiR[kt] = load_b_frag(WiB, kt);
#pragma unroll
  for (int kt = 0; kt < 16; ++kt)  whR[kt] = load_b_frag(WhB, kt);

  // ---- biases are t-invariant: preload this thread's <=3 combine slots ----
  float bR[3], bZ[3], bN[3], bH[3];
#pragma unroll
  for (int k = 0; k < 3; ++k) {
    const int idx = tid + k * NTH;
    if (idx < 512) {
      const int cg = c0 + (idx & 15);
      bR[k] = bi[cg]; bZ[k] = bi[Hc + cg]; bN[k] = bi[2 * Hc + cg]; bH[k] = bhn[cg];
    }
  }

  for (int t = 0; t < Tc; ++t) {
    // ternary (not array) select keeps address-space inference -> global
    const unsigned short* hcur = pr ? hbuf1 : hbuf0;
    unsigned short*       hnxt = pr ? hbuf0 : hbuf1;

    // ---- async stage: h (32 KB, contiguous) and x_t (row slab) -> LDS ----
    for (int i = tid; i < 2048; i += NTH)           // 2048 x 16 B = 32 KB
      async_copy16(hOff + (unsigned)i * 16, hcur + (size_t)i * 8);
    for (int i = tid; i < KTX * 128; i += NTH) {    // 32 rows x KTX*4 chunks
      const int r = i / (KTX * 4);
      const int c = i - r * (KTX * 4);
      async_copy16(xOff + (unsigned)(r * (KTX * 32) + c * 8) * 2,
                   xbase + (size_t)t * (KTX * 32) + (size_t)r * xrs + c * 8);
    }
    wait_async();        // own wave's ASYNCcnt == 0
    __syncthreads();     // all waves' copies complete

    // ---- GEMM from LDS against VGPR-resident weights -----------------------
    const unsigned short* xrow = xTile + m * (KTX * 32) + hf * 8;
    const unsigned short* hrow = hTile + m * Hc + hf * 8;
    v8f acc_x = {};   // input-side accumulation (xi)
    v8f acc_h = {};   // hidden-side accumulation (hh)
#pragma unroll
    for (int kt = 0; kt < KTX; ++kt)
      acc_x = wmma_bf16(load_a_frag(xrow, kt << 5), wiR[kt], acc_x);
#pragma unroll
    for (int kt = 0; kt < 16; ++kt)
      acc_h = wmma_bf16(load_a_frag(hrow, kt << 5), whR[kt], acc_h);

    __syncthreads();  // xTile reads done before sAcc (alias) is written

    // stage tiles into LDS (C/D layout: elem e -> m = mtile*16 + hf*8 + e)
    {
      const int n = lane & 15;
#pragma unroll
      for (int e = 0; e < 8; ++e) {
        const int mm = mtile * 16 + hf * 8 + e;
        if (g == 2) { sAcc[2][mm][n] = acc_x[e]; sAcc[3][mm][n] = acc_h[e]; }
        else        { sAcc[g][mm][n] = acc_x[e] + acc_h[e]; }
      }
    }
    __syncthreads();

    // gate combine + hidden update for our 32x16 slab (h_prev from LDS)
#pragma unroll
    for (int k = 0; k < 3; ++k) {
      const int idx = tid + k * NTH;
      if (idx < 512) {
        const int mm = idx >> 4, c = idx & 15;
        const int cg = c0 + c;
        const float r  = 1.0f / (1.0f + __expf(-(sAcc[0][mm][c] + bR[k])));
        const float z  = 1.0f / (1.0f + __expf(-(sAcc[1][mm][c] + bZ[k])));
        const float nn = tanhf(sAcc[2][mm][c] + bN[k] +
                               r * (sAcc[3][mm][c] + bH[k]));
        const float hp = bf2f(hTile[mm * Hc + cg]);
        const unsigned short hv = f2bf((1.0f - z) * nn + z * hp);
        hnxt[mm * Hc + cg] = hv;
        if (WRITE_OUT) h0out[((size_t)mm * Tc + t) * Hc + cg] = hv;
      }
    }

    ++phase;
    grid_sync(bar, NWG * phase);
    pr ^= 1;
  }
}

// ---------------------------------------------------------------------------
// Persistent GRU kernel: both layers' recurrences + final FC.
// ---------------------------------------------------------------------------
__global__ __launch_bounds__(NTH, 1) void gru_persistent(
    const unsigned short* __restrict__ xbf,   // (B,T,I) bf16 row-major
    unsigned short* __restrict__ h0out,       // (B,T,H) bf16 layer-0 outputs
    unsigned short* __restrict__ hbuf0,       // (B,H) bf16 double buffer
    unsigned short* __restrict__ hbuf1,
    unsigned* __restrict__ bar,
    const unsigned short* __restrict__ WiP0, const unsigned short* __restrict__ WhP0,
    const unsigned short* __restrict__ WiP1, const unsigned short* __restrict__ WhP1,
    const unsigned short* __restrict__ WfcP,
    const float* __restrict__ bi0, const float* __restrict__ bhn0,
    const float* __restrict__ bi1, const float* __restrict__ bhn1,
    const float* __restrict__ bfc,
    float* __restrict__ out) {
  const int wg    = blockIdx.x;      // 0..31, owns h columns [wg*16, wg*16+16)
  const int tid   = threadIdx.x;     // 0..191
  const int wave  = tid >> 5;        // 0..5
  const int lane  = tid & 31;
  const int mtile = wave & 1;        // rows 0-15 / 16-31
  const int g     = wave >> 1;       // 0=r 1=z 2=n
  const int hf    = lane >> 4;
  const int c0    = wg * 16;

  __shared__ __align__(16) unsigned char smem[65536];  // hTile | xTile/sAcc

  unsigned phase = 0;
  int pr = 0;

  // ---- layer 0 ----  (x: row stride T*I, K tiles = I/32 = 8)
  gru_layer<Ic / 32, true>(xbf, (size_t)Tc * Ic, WiP0, WhP0, bi0, bhn0,
                           h0out, hbuf0, hbuf1, pr, phase, bar, smem,
                           tid, lane, mtile, g, hf, c0, wg);

  // h0 = 0 for layer 1: each WG zeroes its own columns of the read buffer
  {
    unsigned short* hrd = pr ? hbuf1 : hbuf0;
    for (int idx = tid; idx < 32 * 16; idx += NTH) {
      const int m = idx >> 4, c = idx & 15;
      hrd[m * Hc + c0 + c] = 0;
    }
  }
  ++phase;
  grid_sync(bar, NWG * phase);

  // ---- layer 1 ----  (x = h0out: row stride T*H, K tiles = H/32 = 16)
  gru_layer<Hc / 32, false>(h0out, (size_t)Tc * Hc, WiP1, WhP1, bi1, bhn1,
                            h0out, hbuf0, hbuf1, pr, phase, bar, smem,
                            tid, lane, mtile, g, hf, c0, wg);

  // ---- final FC: out = h_T @ Wfc + bfc  (32x64) by WG 0, WMMA again ------
  if (wg == 0) {
    const unsigned short* hT = pr ? hbuf1 : hbuf0;
    for (int tile = wave; tile < 8; tile += 6) {   // 2 mtiles x 4 ntiles
      const int mt = tile & 1;
      const int nt = tile >> 1;
      const unsigned short* hrow = hT + (size_t)(mt * 16 + (lane & 15)) * Hc + hf * 8;
      const unsigned short* WfB  = WfcP + ((size_t)(nt * 16) * 32 + lane) * 16;
      v8f acc = {};
#pragma unroll
      for (int kt = 0; kt < 16; ++kt) {
        const u16x8* p = (const u16x8*)(hrow + (kt << 5));
        u16x8 lo = p[0], hi = p[2];
        u16x16 u;
#pragma unroll
        for (int e = 0; e < 8; ++e) { u[e] = lo[e]; u[e + 8] = hi[e]; }
        acc = wmma_bf16(__builtin_bit_cast(v16bf, u), load_b_frag(WfB, kt), acc);
      }
      const int n = nt * 16 + (lane & 15);
#pragma unroll
      for (int e = 0; e < 8; ++e) {
        const int mm = mt * 16 + hf * 8 + e;
        out[mm * Oc + n] = acc[e] + bfc[n];
      }
    }
  }
}

// ---------------------------------------------------------------------------
extern "C" void kernel_launch(void* const* d_in, const int* in_sizes, int n_in,
                              void* d_out, int out_size, void* d_ws, size_t ws_size,
                              hipStream_t stream) {
  (void)in_sizes; (void)n_in; (void)out_size; (void)ws_size;
  const float* x    = (const float*)d_in[0];
  const float* Wi0  = (const float*)d_in[1];
  const float* bi0  = (const float*)d_in[2];
  const float* Wh0  = (const float*)d_in[3];
  const float* bhn0 = (const float*)d_in[4];
  const float* Wi1  = (const float*)d_in[5];
  const float* bi1  = (const float*)d_in[6];
  const float* Wh1  = (const float*)d_in[7];
  const float* bhn1 = (const float*)d_in[8];
  const float* Wfc  = (const float*)d_in[9];
  const float* bfc  = (const float*)d_in[10];
  float* out = (float*)d_out;

  char* ws = (char*)d_ws;
  size_t off = 0;
  auto take = [&](size_t bytes) -> char* {
    char* p = ws + off;
    off += (bytes + 255) & ~(size_t)255;
    return p;
  };

  unsigned*       bar   = (unsigned*)      take(256);
  unsigned short* hbuf0 = (unsigned short*)take((size_t)Bc * Hc * 2);
  unsigned short* hbuf1 = (unsigned short*)take((size_t)Bc * Hc * 2);
  unsigned short* xbf   = (unsigned short*)take((size_t)Bc * Tc * Ic * 2);
  unsigned short* h0out = (unsigned short*)take((size_t)Bc * Tc * Hc * 2);
  unsigned short* WiP0  = (unsigned short*)take((size_t)Ic * 3 * Hc * 2);
  unsigned short* WhP0  = (unsigned short*)take((size_t)Hc * 3 * Hc * 2);
  unsigned short* WiP1  = (unsigned short*)take((size_t)Hc * 3 * Hc * 2);
  unsigned short* WhP1  = (unsigned short*)take((size_t)Hc * 3 * Hc * 2);
  unsigned short* WfcP  = (unsigned short*)take((size_t)Hc * Oc * 2);

  // zero: barrier counter + both h double-buffers (bf16 zero == 0x0000)
  hipMemsetAsync(d_ws, 0, 256 + 2 * ((size_t)Bc * Hc * 2), stream);

  // prep: x -> bf16, weights -> packed bf16 B-fragments
  f32_to_bf16<<<4096, 256, 0, stream>>>(x, xbf, (size_t)Bc * Tc * Ic);
  auto packThreads = [](int K, int N) { return (K >> 5) * (N >> 4) * 32; };
  pack_b<<<(packThreads(Ic, 3 * Hc) + 255) / 256, 256, 0, stream>>>(Wi0, WiP0, Ic, 3 * Hc);
  pack_b<<<(packThreads(Hc, 3 * Hc) + 255) / 256, 256, 0, stream>>>(Wh0, WhP0, Hc, 3 * Hc);
  pack_b<<<(packThreads(Hc, 3 * Hc) + 255) / 256, 256, 0, stream>>>(Wi1, WiP1, Hc, 3 * Hc);
  pack_b<<<(packThreads(Hc, 3 * Hc) + 255) / 256, 256, 0, stream>>>(Wh1, WhP1, Hc, 3 * Hc);
  pack_b<<<(packThreads(Hc, Oc)     + 255) / 256, 256, 0, stream>>>(Wfc, WfcP, Hc, Oc);

  // persistent recurrence + FC
  gru_persistent<<<NWG, NTH, 0, stream>>>(xbf, h0out, hbuf0, hbuf1, bar,
                                          WiP0, WhP0, WiP1, WhP1, WfcP,
                                          bi0, bhn0, bi1, bhn1, bfc, out);
}
